// NSA_21990232555796
// MI455X (gfx1250) — matrix operand order repaired and checked by reading, not compiled, not run on hardware
//
#include <hip/hip_runtime.h>

// ---------------- CDNA5 WMMA types ----------------
typedef __attribute__((ext_vector_type(16))) __bf16 v16bf;
typedef __attribute__((ext_vector_type(8)))  float  v8f;

static __device__ __forceinline__ v8f wmma_bf16(v16bf a, v16bf b, v8f c) {
    // v_wmma_f32_16x16x32_bf16 : D = A(16x32) * B(32x16) + C(16x16)
    return __builtin_amdgcn_wmma_f32_16x16x32_bf16(false, a, false, b, (short)0, c, false, false);
}

// fragment <-> 16B-chunk view for vectorized LDS access (2x ds_load_b128)
union Frag16 { v16bf v; uint4 q[2]; };

// ---------------- TDM (Tensor Data Mover) ----------------
#if defined(__has_builtin)
#  if __has_builtin(__builtin_amdgcn_tensor_load_to_lds) && __has_builtin(__builtin_amdgcn_s_wait_tensorcnt)
#    define NSA_USE_TDM 1
#  endif
#endif
#ifndef NSA_USE_TDM
#  define NSA_USE_TDM 0
#endif

#if NSA_USE_TDM
typedef unsigned int tdm_v4u __attribute__((ext_vector_type(4)));
typedef int          tdm_v8i __attribute__((ext_vector_type(8)));
typedef int          tdm_v4i __attribute__((ext_vector_type(4)));

// loop-invariant part of the D# (group1): 2-byte elements, 2-D tile
struct TdmDesc { tdm_v8i g1; };

static __device__ __forceinline__ TdmDesc tdm_make(unsigned tile_d0, unsigned tile_d1,
                                                   unsigned stride_elems) {
    TdmDesc d;
    d.g1[0] = (int)(1u << 16);                               // data_size = 2 bytes
    d.g1[1] = (int)((tile_d0 & 0xFFFFu) << 16);              // tensor_dim0 lo16 (== tile)
    d.g1[2] = (int)(((tile_d0 >> 16) & 0xFFFFu) | ((tile_d1 & 0xFFFFu) << 16));
    d.g1[3] = (int)(((tile_d1 >> 16) & 0xFFFFu) | (tile_d0 << 16)); // dim1 hi | tile_dim0
    d.g1[4] = (int)(tile_d1 & 0xFFFFu);                      // tile_dim1 (tile_dim2=0)
    d.g1[5] = (int)stride_elems;                             // tensor_dim0_stride lo32
    d.g1[6] = 0;
    d.g1[7] = 0;
    return d;
}

// per-issue: only group0 (lds addr + global addr) changes
static __device__ __forceinline__ void tdm_issue(const TdmDesc& d, unsigned lds_byte_off,
                                                 const void* gptr) {
    unsigned long long ga = (unsigned long long)gptr;
    tdm_v4u g0;
    g0[0] = 1u;                                              // count=1 (valid user D#)
    g0[1] = lds_byte_off;
    g0[2] = (unsigned)ga;
    g0[3] = (unsigned)((ga >> 32) & 0x01FFFFFFu) | (2u << 30); // addr[56:32] | type=2
    tdm_v4i z = {0, 0, 0, 0};
#if __clang_major__ >= 23
    tdm_v8i z8 = {0, 0, 0, 0, 0, 0, 0, 0};
    __builtin_amdgcn_tensor_load_to_lds(g0, d.g1, z, z, z8, 0);
#else
    __builtin_amdgcn_tensor_load_to_lds(g0, d.g1, z, z, 0);
#endif
}
static __device__ __forceinline__ unsigned lds_off(const void* p) {
    return (unsigned)(unsigned long long)p;   // LDS aperture: low 32 bits = LDS byte offset
}
#endif

// ---------------- problem constants ----------------
constexpr int Bz = 2, S = 2048, D = 2048, H = 32, DH = 64;
constexpr int MR = Bz * S;   // 4096 token rows
constexpr int HD = H * DH;   // 2048

// =====================================================================
// fp32 -> bf16 convert
// =====================================================================
__global__ __launch_bounds__(256)
void nsa_cvt_bf16(const float* __restrict__ src, __bf16* __restrict__ dst, int n4) {
    int i = blockIdx.x * 256 + threadIdx.x;
    if (i >= n4) return;
    float4 f = ((const float4*)src)[i];
    __bf16 o[4] = {(__bf16)f.x, (__bf16)f.y, (__bf16)f.z, (__bf16)f.w};
    *(uint2*)&dst[(size_t)i * 4] = *(const uint2*)o;
}

// =====================================================================
// W[K,N] fp32 -> WT[N,K] bf16 (tiled transpose through LDS)
// =====================================================================
__global__ __launch_bounds__(256)
void nsa_cvt_transpose(const float* __restrict__ W, __bf16* __restrict__ WT, int K, int N) {
    __shared__ float T[32][33];
    int n0 = blockIdx.x * 32, k0 = blockIdx.y * 32;
    for (int r = threadIdx.y; r < 32; r += 8)
        T[r][threadIdx.x] = W[(size_t)(k0 + r) * N + n0 + threadIdx.x];
    __syncthreads();
    for (int r = threadIdx.y; r < 32; r += 8)
        WT[(size_t)(n0 + r) * K + k0 + threadIdx.x] = (__bf16)T[threadIdx.x][r];
}

// =====================================================================
// GEMM: C[M,N] = A[M,K](bf16) @ WT[N,K](bf16 pre-transposed) + bias
// Block 64x128, 8 waves (2x4), wave 32x32, K-step 64 (8 WMMA / round).
// Peeled double-buffered TDM pipeline (unconditional steady state).
// =====================================================================
__global__ __launch_bounds__(256)
void nsa_gemm_bias(const __bf16* __restrict__ A, const __bf16* __restrict__ WT,
                   const float* __restrict__ bias,
                   float* __restrict__ Cf, __bf16* __restrict__ Cb,
                   int M, int N, int K) {
    __shared__ __bf16 As[2][64][64];
    __shared__ __bf16 BsT[2][128][64];

    const int tid  = threadIdx.x;
    const int lane = tid & 31, wave = tid >> 5;
    const int half = lane >> 4, l16 = lane & 15;
    const int waveM = wave >> 2, waveN = wave & 3;
    const int blockM = blockIdx.y * 64;
    const int blockN = blockIdx.x * 128;
    const int nk = K / 64;

    v8f c[2][2] = {};

    auto compute = [&](int bf) {
        #pragma unroll
        for (int sub = 0; sub < 64; sub += 32) {
            Frag16 a[2], b[2];
            #pragma unroll
            for (int mi = 0; mi < 2; ++mi) {
                int row = waveM * 32 + mi * 16 + l16;
                a[mi].q[0] = *(const uint4*)&As[bf][row][sub + 8 * half];
                a[mi].q[1] = *(const uint4*)&As[bf][row][sub + 16 + 8 * half];
            }
            #pragma unroll
            for (int ni = 0; ni < 2; ++ni) {
                int col = waveN * 32 + ni * 16 + l16;
                b[ni].q[0] = *(const uint4*)&BsT[bf][col][sub + 8 * half];
                b[ni].q[1] = *(const uint4*)&BsT[bf][col][sub + 16 + 8 * half];
            }
            c[0][0] = wmma_bf16(a[0].v, b[0].v, c[0][0]);
            c[0][1] = wmma_bf16(a[0].v, b[1].v, c[0][1]);
            c[1][0] = wmma_bf16(a[1].v, b[0].v, c[1][0]);
            c[1][1] = wmma_bf16(a[1].v, b[1].v, c[1][1]);
        }
    };

#if NSA_USE_TDM
    const TdmDesc dA = tdm_make(64, 64,  (unsigned)K);
    const TdmDesc dB = tdm_make(64, 128, (unsigned)K);
    if (wave == 0) {
        tdm_issue(dA, lds_off(&As[0][0][0]),  A  + (size_t)blockM * K);
        tdm_issue(dB, lds_off(&BsT[0][0][0]), WT + (size_t)blockN * K);
    }
    int buf = 0;
    for (int it = 0; it < nk - 1; ++it, buf ^= 1) {
        if (wave == 0) {   // unconditional steady-state issue of tile it+1
            tdm_issue(dA, lds_off(&As[buf ^ 1][0][0]),
                      A + (size_t)blockM * K + (it + 1) * 64);
            tdm_issue(dB, lds_off(&BsT[buf ^ 1][0][0]),
                      WT + (size_t)blockN * K + (it + 1) * 64);
            __builtin_amdgcn_s_wait_tensorcnt(2);   // tile it landed
        }
        __syncthreads();
        compute(buf);
        __syncthreads();   // tile consumed -> its buffer reusable
    }
    if (wave == 0) __builtin_amdgcn_s_wait_tensorcnt(0);
    __syncthreads();
    compute(buf);          // final tile
#else
    for (int it = 0; it < nk; ++it) {
        __syncthreads();
        for (int i = tid; i < 64 * 64 / 8; i += 256) {
            int r = i >> 3, cc = (i & 7) * 8;
            *(uint4*)&As[0][r][cc] = *(const uint4*)&A[(size_t)(blockM + r) * K + it * 64 + cc];
        }
        for (int i = tid; i < 128 * 64 / 8; i += 256) {
            int r = i >> 3, cc = (i & 7) * 8;
            *(uint4*)&BsT[0][r][cc] = *(const uint4*)&WT[(size_t)(blockN + r) * K + it * 64 + cc];
        }
        __syncthreads();
        compute(0);
    }
#endif

    #pragma unroll
    for (int mi = 0; mi < 2; ++mi)
        #pragma unroll
        for (int ni = 0; ni < 2; ++ni)
            #pragma unroll
            for (int j = 0; j < 8; ++j) {
                int row = blockM + waveM * 32 + mi * 16 + j + 8 * half;
                int col = blockN + waveN * 32 + ni * 16 + l16;
                float vv = c[mi][ni][j] + bias[col];
                if (Cb) Cb[(size_t)row * N + col] = (__bf16)vv;
                else    Cf[(size_t)row * N + col] = vv;
            }
}

// =====================================================================
// gate g[row] = sigmoid((x[row,:] . Wg) * scaling)   (wave per row)
// =====================================================================
__global__ __launch_bounds__(256)
void nsa_gate(const float* __restrict__ x, const float* __restrict__ Wg,
              const float* __restrict__ scaling, float* __restrict__ g,
              int rows, int K) {
    int w = (int)((blockIdx.x * blockDim.x + threadIdx.x) >> 5);
    int lane = threadIdx.x & 31;
    if (w >= rows) return;
    const float* xr = x + (size_t)w * K;
    float acc = 0.f;
    for (int k = lane; k < K; k += 32) acc += xr[k] * Wg[k];
    #pragma unroll
    for (int d = 16; d >= 1; d >>= 1) acc += __shfl_xor(acc, d, 32);
    if (lane == 0) g[w] = 1.f / (1.f + __expf(-acc * scaling[0]));
}

// =====================================================================
// compress: y = LN(relu(row @ Wc + bc)) per (token, head); wave per job.
// vt=0 -> out [B,H,S,DH] (kc); vt=1 -> out transposed [B,H,DH,S] (vc).
// =====================================================================
__global__ __launch_bounds__(256)
void nsa_compress(const __bf16* __restrict__ kv,   // [MR, HD] bf16
                  const float* __restrict__ Wc, const float* __restrict__ bc,
                  const float* __restrict__ gamma, const float* __restrict__ beta,
                  __bf16* __restrict__ outc, int vt) {
    int gw = (int)((blockIdx.x * blockDim.x + threadIdx.x) >> 5);
    int lane = threadIdx.x & 31;
    int h = gw & (H - 1);
    int rs = gw >> 5;                 // token row (H == 32)
    if (rs >= MR) return;
    const __bf16* row = kv + (size_t)rs * HD + h * DH;

    float v0 = bc[lane], v1 = bc[lane + 32];
    #pragma unroll 4
    for (int k = 0; k < DH; ++k) {
        float rk = (float)row[k];
        v0 += rk * Wc[k * DH + lane];
        v1 += rk * Wc[k * DH + lane + 32];
    }
    v0 = fmaxf(v0, 0.f);
    v1 = fmaxf(v1, 0.f);

    float sum = v0 + v1;
    #pragma unroll
    for (int d = 16; d >= 1; d >>= 1) sum += __shfl_xor(sum, d, 32);
    float mean = sum * (1.f / 64.f);
    float d0 = v0 - mean, d1 = v1 - mean;
    float vs = d0 * d0 + d1 * d1;
    #pragma unroll
    for (int d = 16; d >= 1; d >>= 1) vs += __shfl_xor(vs, d, 32);
    float rstd = rsqrtf(vs * (1.f / 64.f) + 1e-5f);

    int b = rs / S, s = rs - b * S;
    float y0 = d0 * rstd * gamma[lane] + beta[lane];
    float y1 = d1 * rstd * gamma[lane + 32] + beta[lane + 32];
    if (vt) {   // [B,H,DH,S]
        size_t ob = (size_t)(b * H + h) * DH * S;
        outc[ob + (size_t)lane * S + s]        = (__bf16)y0;
        outc[ob + (size_t)(lane + 32) * S + s] = (__bf16)y1;
    } else {    // [B,H,S,DH]
        size_t ob = ((size_t)(b * H + h) * S + s) * DH;
        outc[ob + lane]      = (__bf16)y0;
        outc[ob + lane + 32] = (__bf16)y1;
    }
}

// =====================================================================
// gated flash attention. Block = 128 thr (4 waves) per (b,h,64 q rows).
// 64-key tiles: 16 WMMA per barrier round. Peeled TDM double buffer.
// =====================================================================
__global__ __launch_bounds__(128)
void nsa_attn(const __bf16* __restrict__ qbuf,   // [MR, HD] bf16
              const __bf16* __restrict__ kc,     // [B,H,S,DH]
              const __bf16* __restrict__ vcT,    // [B,H,DH,S]
              const float* __restrict__ gbuf,    // [MR]
              __bf16* __restrict__ obuf) {       // [MR, HD] bf16
    __shared__ __bf16 Qs[64][64];
    __shared__ __bf16 Ks[2][64][64];     // [key][d]
    __shared__ __bf16 VsT[2][64][64];    // [d][key]
    __shared__ __bf16 Ps[4][16][64];

    const int tid = threadIdx.x, lane = tid & 31, wave = tid >> 5;
    const int half = lane >> 4, l16 = lane & 15;
    const int q0 = blockIdx.x * 64;
    const int h = blockIdx.y, b = blockIdx.z;
    constexpr int NT = S / 64;

    const __bf16* kcb = kc  + (size_t)(b * H + h) * S * DH;
    const __bf16* vcb = vcT + (size_t)(b * H + h) * DH * S;
    const __bf16* qg  = qbuf + (size_t)(b * S + q0) * HD + h * DH;

    float gr[8];
    #pragma unroll
    for (int j = 0; j < 8; ++j)
        gr[j] = gbuf[b * S + q0 + wave * 16 + j + 8 * half];

    float m[8], l[8];
    v8f acc[4] = {};
    #pragma unroll
    for (int j = 0; j < 8; ++j) { m[j] = -3.0e38f; l[j] = 0.f; }

    Frag16 qa[2];
    auto hoist_q = [&]() {
        int row = wave * 16 + l16;
        #pragma unroll
        for (int t = 0; t < 2; ++t) {
            qa[t].q[0] = *(const uint4*)&Qs[row][t * 32 + 8 * half];
            qa[t].q[1] = *(const uint4*)&Qs[row][t * 32 + 16 + 8 * half];
        }
    };

    auto compute = [&](int bf) {
        // scores S = Q @ Kc^T : four 16x16 c-tiles over 64 keys
        v8f s[4] = {};
        #pragma unroll
        for (int t = 0; t < 2; ++t)
            #pragma unroll
            for (int n = 0; n < 4; ++n) {
                Frag16 bk;
                bk.q[0] = *(const uint4*)&Ks[bf][n * 16 + l16][t * 32 + 8 * half];
                bk.q[1] = *(const uint4*)&Ks[bf][n * 16 + l16][t * 32 + 16 + 8 * half];
                s[n] = wmma_bf16(qa[t].v, bk.v, s[n]);
            }
        // gated online softmax (rows per-j within 16-lane half)
        #pragma unroll
        for (int j = 0; j < 8; ++j) {
            float sj[4];
            #pragma unroll
            for (int n = 0; n < 4; ++n) sj[n] = s[n][j] * gr[j];
            float mx = fmaxf(fmaxf(sj[0], sj[1]), fmaxf(sj[2], sj[3]));
            #pragma unroll
            for (int d = 8; d >= 1; d >>= 1) mx = fmaxf(mx, __shfl_xor(mx, d, 32));
            float mn = fmaxf(m[j], mx);
            float alpha = __expf(m[j] - mn);
            float rs = 0.f;
            #pragma unroll
            for (int n = 0; n < 4; ++n) {
                float p = __expf(sj[n] - mn);
                Ps[wave][j + 8 * half][n * 16 + l16] = (__bf16)p;
                rs += p;
            }
            #pragma unroll
            for (int d = 8; d >= 1; d >>= 1) rs += __shfl_xor(rs, d, 32);
            l[j] = l[j] * alpha + rs;
            m[j] = mn;
            #pragma unroll
            for (int n = 0; n < 4; ++n) acc[n][j] *= alpha;
        }
        // acc += P @ Vc (2 k-steps x 4 d-tiles)
        #pragma unroll
        for (int t = 0; t < 2; ++t) {
            Frag16 pa;
            pa.q[0] = *(const uint4*)&Ps[wave][l16][t * 32 + 8 * half];
            pa.q[1] = *(const uint4*)&Ps[wave][l16][t * 32 + 16 + 8 * half];
            #pragma unroll
            for (int n = 0; n < 4; ++n) {
                Frag16 bv;
                bv.q[0] = *(const uint4*)&VsT[bf][n * 16 + l16][t * 32 + 8 * half];
                bv.q[1] = *(const uint4*)&VsT[bf][n * 16 + l16][t * 32 + 16 + 8 * half];
                acc[n] = wmma_bf16(pa.v, bv.v, acc[n]);
            }
        }
    };

#if NSA_USE_TDM
    const TdmDesc dQ = tdm_make(64, 64, (unsigned)HD);
    const TdmDesc dK = tdm_make(4096, 1, 4096);        // contiguous 64x64 tile
    const TdmDesc dV = tdm_make(64, 64, (unsigned)S);
    if (wave == 0) {
        tdm_issue(dQ, lds_off(&Qs[0][0]),     qg);
        tdm_issue(dK, lds_off(&Ks[0][0][0]),  kcb);
        tdm_issue(dV, lds_off(&VsT[0][0][0]), vcb);
        __builtin_amdgcn_s_wait_tensorcnt(2);          // Q landed (in-order)
    }
    __syncthreads();
    hoist_q();
    int buf = 0;
    for (int kt = 0; kt < NT - 1; ++kt, buf ^= 1) {
        if (wave == 0) {
            tdm_issue(dK, lds_off(&Ks[buf ^ 1][0][0]),  kcb + (size_t)(kt + 1) * 64 * DH);
            tdm_issue(dV, lds_off(&VsT[buf ^ 1][0][0]), vcb + (size_t)(kt + 1) * 64);
            __builtin_amdgcn_s_wait_tensorcnt(2);      // tile kt landed
        }
        __syncthreads();
        compute(buf);
        __syncthreads();
    }
    if (wave == 0) __builtin_amdgcn_s_wait_tensorcnt(0);
    __syncthreads();
    compute(buf);
#else
    for (int i = tid; i < 64 * 64 / 8; i += 128) {
        int r = i >> 3, cc = (i & 7) * 8;
        *(uint4*)&Qs[r][cc] = *(const uint4*)&qg[(size_t)r * HD + cc];
    }
    __syncthreads();
    hoist_q();
    for (int kt = 0; kt < NT; ++kt) {
        __syncthreads();
        for (int i = tid; i < 64 * 64 / 8; i += 128) {
            int r = i >> 3, cc = (i & 7) * 8;
            *(uint4*)&Ks[0][r][cc]  = *(const uint4*)&kcb[(size_t)(kt * 64 + r) * DH + cc];
            *(uint4*)&VsT[0][r][cc] = *(const uint4*)&vcb[(size_t)r * S + kt * 64 + cc];
        }
        __syncthreads();
        compute(0);
    }
#endif

    #pragma unroll
    for (int n = 0; n < 4; ++n)
        #pragma unroll
        for (int j = 0; j < 8; ++j) {
            int row = q0 + wave * 16 + j + 8 * half;
            int col = n * 16 + l16;
            obuf[(size_t)(b * S + row) * HD + h * DH + col] = (__bf16)(acc[n][j] / l[j]);
        }
}

// =====================================================================
// launch
// =====================================================================
extern "C" void kernel_launch(void* const* d_in, const int* in_sizes, int n_in,
                              void* d_out, int out_size, void* d_ws, size_t ws_size,
                              hipStream_t stream) {
    const float* x    = (const float*)d_in[0];
    const float* Wq   = (const float*)d_in[1];
    const float* bq   = (const float*)d_in[2];
    const float* Wk   = (const float*)d_in[3];
    const float* bk   = (const float*)d_in[4];
    const float* Wv   = (const float*)d_in[5];
    const float* bv   = (const float*)d_in[6];
    const float* Wo   = (const float*)d_in[7];
    const float* bo   = (const float*)d_in[8];
    const float* Wg   = (const float*)d_in[9];
    const float* scal = (const float*)d_in[10];
    const float* Wck  = (const float*)d_in[11];
    const float* bck  = (const float*)d_in[12];
    const float* gck  = (const float*)d_in[13];
    const float* beck = (const float*)d_in[14];
    const float* Wcv  = (const float*)d_in[15];
    const float* bcv  = (const float*)d_in[16];
    const float* gcv  = (const float*)d_in[17];
    const float* becv = (const float*)d_in[18];

    char* ws = (char*)d_ws;
    const size_t bfmat = (size_t)MR * HD * sizeof(__bf16);   // 16.8 MB
    __bf16* xbf  = (__bf16*)(ws + 0 * bfmat);
    __bf16* WTq  = (__bf16*)(ws + 1 * bfmat);
    __bf16* WTk  = (__bf16*)(ws + 2 * bfmat);
    __bf16* WTv  = (__bf16*)(ws + 3 * bfmat);
    __bf16* WTo  = (__bf16*)(ws + 4 * bfmat);
    __bf16* qbuf = (__bf16*)(ws + 5 * bfmat);
    __bf16* kbuf = (__bf16*)(ws + 6 * bfmat);
    __bf16* vbuf = (__bf16*)(ws + 7 * bfmat);
    __bf16* obuf = (__bf16*)(ws + 8 * bfmat);
    __bf16* kcb  = (__bf16*)(ws + 9 * bfmat);
    __bf16* vcb  = (__bf16*)(ws + 10 * bfmat);
    float*  gbuf = (float*) (ws + 11 * bfmat);

    // convert once: x -> bf16, weights -> bf16 transposed
    nsa_cvt_bf16<<<(MR * D / 4 + 255) / 256, 256, 0, stream>>>(x, xbf, MR * D / 4);
    dim3 tGrid(HD / 32, D / 32), tBlk(32, 8);
    nsa_cvt_transpose<<<tGrid, tBlk, 0, stream>>>(Wq, WTq, D, HD);
    nsa_cvt_transpose<<<tGrid, tBlk, 0, stream>>>(Wk, WTk, D, HD);
    nsa_cvt_transpose<<<tGrid, tBlk, 0, stream>>>(Wv, WTv, D, HD);
    nsa_cvt_transpose<<<dim3(D / 32, HD / 32), tBlk, 0, stream>>>(Wo, WTo, HD, D);

    // Q, K, V projections (bf16 out)
    dim3 gGrid(HD / 128, MR / 64);
    nsa_gemm_bias<<<gGrid, 256, 0, stream>>>(xbf, WTq, bq, nullptr, qbuf, MR, HD, D);
    nsa_gemm_bias<<<gGrid, 256, 0, stream>>>(xbf, WTk, bk, nullptr, kbuf, MR, HD, D);
    nsa_gemm_bias<<<gGrid, 256, 0, stream>>>(xbf, WTv, bv, nullptr, vbuf, MR, HD, D);
    // gate
    nsa_gate<<<(MR * 32 + 255) / 256, 256, 0, stream>>>(x, Wg, scal, gbuf, MR, D);
    // K/V compression (wave per token-head job); vc emitted transposed
    int cjobs = MR * H;
    nsa_compress<<<(cjobs * 32 + 255) / 256, 256, 0, stream>>>(kbuf, Wck, bck, gck, beck, kcb, 0);
    nsa_compress<<<(cjobs * 32 + 255) / 256, 256, 0, stream>>>(vbuf, Wcv, bcv, gcv, becv, vcb, 1);
    // attention
    dim3 aGrid(S / 64, H, Bz);
    nsa_attn<<<aGrid, 128, 0, stream>>>(qbuf, kcb, vcb, gbuf, obuf);
    // output projection (fp32 out)
    dim3 oGrid(D / 128, MR / 64);
    nsa_gemm_bias<<<oGrid, 256, 0, stream>>>(obuf, WTo, bo, (float*)d_out, nullptr, MR, D, HD);
}
